// LocalAttention_13125420057375
// MI455X (gfx1250) — compile-verified, matrix operand order checked by reading, not run on hardware
//
#include <hip/hip_runtime.h>
#include <hip/hip_bf16.h>
#include <stdint.h>

// ---------------------------------------------------------------------------
// Windowed local attention for MI455X (gfx1250), wave32 + WMMA bf16.
// B=8, C=192, H=W=256, window 8x8 -> 8192 windows of [C=192, S=64].
// One workgroup (256 threads = 8 waves) per window; all four GEMMs run on
// v_wmma_f32_16x16x32_bf16 with f32 accumulation.
// ---------------------------------------------------------------------------

typedef __attribute__((ext_vector_type(16))) __bf16 v16bf;
typedef __attribute__((ext_vector_type(8)))  float  v8f;

union Frag {
    v16bf   v;
    uint32_t u[8];
};

__device__ __forceinline__ uint16_t f2bf(float f) {
    uint32_t u = __float_as_uint(f);
    uint32_t r = (u + 0x7FFFu + ((u >> 16) & 1u)) >> 16;  // round-to-nearest-even
    return (uint16_t)r;
}

// K offset of packed bf16 pair i (0..7) for half h (lane>>4), per CDNA5 ISA
// 16-bit A-matrix 16x32 layout: VGPR0..3 -> K {0..7}+8h, VGPR4..7 -> K {16..23}+8h.
__device__ __forceinline__ int kpair(int i, int h) {
    return ((i < 4) ? (2 * i) : (16 + 2 * (i - 4))) + 8 * h;
}

// Fragment whose K-pairs are contiguous in memory: matrix stored row-major
// [row][k] (for A) or as B^T row-major [col][k] (for B). Emits b32 loads.
template <typename P>
__device__ __forceinline__ Frag load_frag_contig(const P* p, int ld, int row, int k0, int h) {
    Frag f;
#pragma unroll
    for (int i = 0; i < 8; ++i) {
        int kk = k0 + kpair(i, h);
        f.u[i] = *reinterpret_cast<const uint32_t*>(p + row * ld + kk);
    }
    return f;
}

// Fragment whose K runs across rows (stride ld): pack two u16 loads per pair.
__device__ __forceinline__ Frag load_frag_strided(const uint16_t* p, int ld, int col, int k0, int h) {
    Frag f;
#pragma unroll
    for (int i = 0; i < 8; ++i) {
        int kk = k0 + kpair(i, h);
        uint32_t lo = p[(kk)     * ld + col];
        uint32_t hi = p[(kk + 1) * ld + col];
        f.u[i] = lo | (hi << 16);
    }
    return f;
}

#define WMMA_BF16(A, B, C) \
    __builtin_amdgcn_wmma_f32_16x16x32_bf16(false, (A), false, (B), (short)0, (C), false, false)

// LDS strides (padded off powers of two to dodge bank conflicts)
#define LDX 200   // x window, token-major [s][c], c<192
#define LDQ 584   // qkv, token-major [s][o], o<576
#define LDA 72    // attn [s][t], t<64
#define LDO 200   // attn output, token-major [s][c]

// region sizes in uint16 units
#define REG1_U16 13824              // max(64*LDX, 64*LDA*2(f32) + 64*LDA)
#define REGQ_U16 (64 * LDQ)         // 37376
#define REGO_U16 (64 * LDO)         // 12800
#define SMEM_BYTES ((REG1_U16 + REGQ_U16 + REGO_U16) * 2)   // 128000 B

__global__ void __launch_bounds__(256)
convert_weights_kernel(const float* __restrict__ wq_f, const float* __restrict__ wp_f,
                       uint16_t* __restrict__ wq_bf, uint16_t* __restrict__ wp_bf) {
    int i = blockIdx.x * 256 + threadIdx.x;
    if (i < 576 * 192) wq_bf[i] = f2bf(wq_f[i]);
    if (i < 192 * 192) wp_bf[i] = f2bf(wp_f[i]);
}

__global__ void __launch_bounds__(256)
local_attn_kernel(const float* __restrict__ x,
                  const uint16_t* __restrict__ wq,   // [576][192] bf16
                  const float* __restrict__ bq,      // [576]
                  const uint16_t* __restrict__ wp,   // [192][192] bf16
                  const float* __restrict__ bp,      // [192]
                  float* __restrict__ out) {
    extern __shared__ uint16_t smem[];
    uint16_t* ldsX  = smem;                              // [64][LDX] bf16 window (token-major)
    uint16_t* ldsQ  = smem + REG1_U16;                   // [64][LDQ] bf16 qkv (token-major)
    uint16_t* ldsO  = smem + REG1_U16 + REGQ_U16;        // [64][LDO] bf16 v@attn^T (token-major)
    float*    attnF = reinterpret_cast<float*>(smem);    // [64][LDA] f32 logits (reuses ldsX)
    uint16_t* attnB = smem + 64 * LDA * 2;               // [64][LDA] bf16 softmax(attn)

    const int tid  = threadIdx.x;
    const int lane = tid & 31;
    const int wave = tid >> 5;
    const int lh   = lane >> 4;   // half (0/1)
    const int ln   = lane & 15;

    const int n  = blockIdx.x;    // window id
    const int b  = n >> 10;
    const int hn = (n >> 5) & 31;
    const int wn = n & 31;

    const float* xwin = x + ((size_t)(b * 192) * 256 + hn * 8) * 256 + wn * 8;

    // ---- Phase 1: load window, f32 -> bf16, token-major in LDS --------------
    for (int e = tid; e < 192 * 64; e += 256) {
        int c = e >> 6;
        int s = e & 63;
        float v = xwin[((size_t)c * 256 + (s >> 3)) * 256 + (s & 7)];
        ldsX[s * LDX + c] = f2bf(v);
    }
    __syncthreads();

    // ---- Phase 2: QKV = w_qkv[576x192] @ x[192x64] + b ----------------------
    // 36 M-tiles x 4 N-tiles = 144 tiles, K = 192 (6 wmma steps). 18 tiles/wave.
    for (int t = wave; t < 144; t += 8) {
        int mt = t >> 2, nt = t & 3;
        int arow = mt * 16 + ln;        // output channel for A rows
        int col  = nt * 16 + ln;        // token column
        v8f acc = {};
#pragma unroll
        for (int k0 = 0; k0 < 192; k0 += 32) {
            Frag a  = load_frag_contig(wq, 192, arow, k0, lh);     // weights (L2-resident)
            Frag bf = load_frag_contig(ldsX, LDX, col, k0, lh);    // x^T rows = B cols
            acc = WMMA_BF16(a.v, bf.v, acc);
        }
#pragma unroll
        for (int r = 0; r < 8; ++r) {
            int orow = mt * 16 + lh * 8 + r;
            ldsQ[col * LDQ + orow] = f2bf(acc[r] + bq[orow]);
        }
    }
    __syncthreads();

    // ---- Phase 3: logits = (q^T k) * scale, [64x64], K = 192 ----------------
    const float scale = 0.07216878364870322f;   // 192^-0.5
    for (int t = wave; t < 16; t += 8) {
        int mt = t >> 2, nt = t & 3;
        int srow = mt * 16 + ln;        // query token
        int col  = nt * 16 + ln;        // key token
        v8f acc = {};
#pragma unroll
        for (int k0 = 0; k0 < 192; k0 += 32) {
            Frag a  = load_frag_contig(ldsQ,       LDQ, srow, k0, lh);  // q^T[s][c]
            Frag bf = load_frag_contig(ldsQ + 192, LDQ, col,  k0, lh);  // k^T[t][c]
            acc = WMMA_BF16(a.v, bf.v, acc);
        }
#pragma unroll
        for (int r = 0; r < 8; ++r)
            attnF[(mt * 16 + lh * 8 + r) * LDA + col] = acc[r] * scale;
    }
    __syncthreads();

    // ---- Phase 4: row softmax over t (64 rows; threads 0..63) ---------------
    if (tid < 64) {
        float* rowp = attnF + tid * LDA;
        float mx = rowp[0];
#pragma unroll 8
        for (int t = 1; t < 64; ++t) mx = fmaxf(mx, rowp[t]);
        float sum = 0.f;
#pragma unroll 8
        for (int t = 0; t < 64; ++t) { float e = __expf(rowp[t] - mx); rowp[t] = e; sum += e; }
        float inv = 1.0f / sum;
        uint16_t* ob = attnB + tid * LDA;
#pragma unroll 8
        for (int t = 0; t < 64; ++t) ob[t] = f2bf(rowp[t] * inv);
    }
    __syncthreads();

    // ---- Phase 5: o = v[192x64] @ attn^T[64x64], K = 64 ---------------------
    for (int t = wave; t < 48; t += 8) {
        int mt = t >> 2, nt = t & 3;
        int crow = mt * 16 + ln;        // value channel
        int col  = nt * 16 + ln;        // query token
        v8f acc = {};
#pragma unroll
        for (int k0 = 0; k0 < 64; k0 += 32) {
            Frag a  = load_frag_strided(ldsQ + 384, LDQ, crow, k0, lh); // v[c][t] across rows
            Frag bf = load_frag_contig(attnB, LDA, col, k0, lh);        // attn[s][t] contiguous
            acc = WMMA_BF16(a.v, bf.v, acc);
        }
#pragma unroll
        for (int r = 0; r < 8; ++r) {
            int orow = mt * 16 + lh * 8 + r;
            ldsO[col * LDO + orow] = f2bf(acc[r]);
        }
    }
    __syncthreads();

    // ---- Phase 6: y = w_proj[192x192] @ o[192x64] + b, window-reverse store -
    float* gout = out + ((size_t)(b * 192) * 256 + hn * 8) * 256 + wn * 8;
    for (int t = wave; t < 48; t += 8) {
        int mt = t >> 2, nt = t & 3;
        int arow = mt * 16 + ln;
        int col  = nt * 16 + ln;
        v8f acc = {};
#pragma unroll
        for (int k0 = 0; k0 < 192; k0 += 32) {
            Frag a  = load_frag_contig(wp, 192, arow, k0, lh);
            Frag bf = load_frag_contig(ldsO, LDO, col, k0, lh);
            acc = WMMA_BF16(a.v, bf.v, acc);
        }
        int sy = col >> 3, sx = col & 7;
#pragma unroll
        for (int r = 0; r < 8; ++r) {
            int orow = mt * 16 + lh * 8 + r;
            gout[((size_t)orow * 256 + sy) * 256 + sx] = acc[r] + bp[orow];
        }
    }
}

extern "C" void kernel_launch(void* const* d_in, const int* in_sizes, int n_in,
                              void* d_out, int out_size, void* d_ws, size_t ws_size,
                              hipStream_t stream) {
    const float* x      = (const float*)d_in[0];   // [8,192,256,256]
    const float* w_qkv  = (const float*)d_in[1];   // [576,192]
    const float* b_qkv  = (const float*)d_in[2];   // [576]
    const float* w_proj = (const float*)d_in[3];   // [192,192]
    const float* b_proj = (const float*)d_in[4];   // [192]
    float* out = (float*)d_out;

    uint16_t* wq_bf = (uint16_t*)d_ws;             // 576*192 bf16
    uint16_t* wp_bf = wq_bf + 576 * 192;           // 192*192 bf16

    convert_weights_kernel<<<(576 * 192 + 255) / 256, 256, 0, stream>>>(
        w_qkv, w_proj, wq_bf, wp_bf);

    local_attn_kernel<<<8192, 256, SMEM_BYTES, stream>>>(
        x, wq_bf, b_qkv, wp_bf, b_proj, out);
}
